// PyramidpoolingMamba_7086696038762
// MI455X (gfx1250) — matrix operand order repaired and checked by reading, not compiled, not run on hardware
//
#include <hip/hip_runtime.h>
#include <hip/hip_bf16.h>

typedef __attribute__((ext_vector_type(16))) _Float16 v16h;
typedef __attribute__((ext_vector_type(4)))  _Float16 h4;
typedef __attribute__((ext_vector_type(8)))  float    v8f;

#define B_    8
#define DIM_  128
#define H_    64
#define W_    64
#define HW_   4096
#define L_    1344      // 64 + 256 + 1024
#define DIN_  256
#define DTRANK_ 8
#define DSTATE_ 16

static __device__ __forceinline__ float silu_f(float v) {
  return v / (1.0f + __expf(-v));
}

static __device__ __forceinline__ h4 cvt4(float4 v) {
  h4 h;
  h[0] = (_Float16)v.x; h[1] = (_Float16)v.y;
  h[2] = (_Float16)v.z; h[3] = (_Float16)v.w;
  return h;
}

// ---------------------------------------------------------------------------
// GEMM 1: D[m,n] = sum_k X[m*K+k] * Wt[n*K+k]   (token-major GEMMs)
// Requirements: K % 64 == 0, M % 32 == 0 (true for all call sites).
// Block 128 threads = 4 waves; block tile 32(M) x 64(N); K-step 64.
// Wave tile 16(M) x 32(N): 2 accumulators, 4 WMMAs per K-step.
// float4 global loads -> packed f16 ds_store_b64; double-buffered LDS with a
// single barrier per K-step; next tile's global loads issued before compute.
// ---------------------------------------------------------------------------
__global__ __launch_bounds__(128)
void gemm_xw_t(const float* __restrict__ X, const float* __restrict__ Wt,
               float* __restrict__ Dout, int M, int N, int K) {
  __shared__ _Float16 As[2][32 * 64];   // 32 m-rows x 64 k
  __shared__ _Float16 Bs[2][64 * 64];   // 64 n-rows x 64 k
  const int tid  = threadIdx.x;
  const int lane = tid & 31;
  const int wave = tid >> 5;
  const int m0 = blockIdx.y * 32;
  const int n0 = blockIdx.x * 64;
  const int miw = (wave >> 1) << 4;    // wave m offset within tile (0/16)
  const int niw = (wave & 1) << 5;     // wave n offset within tile (0/32)

  float4 ra[4];   // A staging: 2048 elems / 128 thr / 4 per vec
  float4 rb[8];   // B staging: 4096 elems / 128 thr / 4 per vec

  auto loadA = [&](int k0) {           // unguarded: M, K tile-aligned
#pragma unroll
    for (int j = 0; j < 4; ++j) {
      int i4 = tid + j * 128;          // 16 float4 per 64-wide row
      int r = i4 >> 4, kq = (i4 & 15) << 2;
      ra[j] = *(const float4*)&X[(size_t)(m0 + r) * K + k0 + kq];
    }
  };
  auto loadB = [&](int k0) {           // row guard only (N=40 tile)
#pragma unroll
    for (int j = 0; j < 8; ++j) {
      int i4 = tid + j * 128;
      int r = i4 >> 4, kq = (i4 & 15) << 2;
      if (n0 + r < N) {
        rb[j] = *(const float4*)&Wt[(size_t)(n0 + r) * K + k0 + kq];
      } else {
        rb[j] = make_float4(0.0f, 0.0f, 0.0f, 0.0f);
      }
    }
  };
  auto storeAB = [&](int buf) {
#pragma unroll
    for (int j = 0; j < 4; ++j)
      *(h4*)&As[buf][(tid + j * 128) << 2] = cvt4(ra[j]);
#pragma unroll
    for (int j = 0; j < 8; ++j)
      *(h4*)&Bs[buf][(tid + j * 128) << 2] = cvt4(rb[j]);
  };

  v8f acc0 = {}, acc1 = {};
  const int arow = miw + (lane & 15);       // A row within 32-row tile
  const int ka = (lane >> 4) << 3;          // +8 upper lane half (A frag)
  const int kb = (lane >> 4) << 4;          // +16 upper lane half (unused; B frag uses ka)
  const int ncl = lane & 15;

  loadA(0); loadB(0);
  storeAB(0);
  __syncthreads();

  int buf = 0;
  for (int k0 = 0; k0 < K; k0 += 64) {
    const bool more = (k0 + 64) < K;
    if (more) {          // next-tile global loads in flight during WMMAs
      loadA(k0 + 64);
      loadB(k0 + 64);
    }
    const _Float16* Ab = As[buf];
    const _Float16* Bb = Bs[buf];
#pragma unroll
    for (int sub = 0; sub < 64; sub += 32) {
      v16h a, b0, b1;
#pragma unroll
      for (int vv = 0; vv < 8; ++vv) {
        int kbase = sub + ((vv < 4) ? (2 * vv) : (16 + 2 * (vv - 4))) + ka;
        a[2 * vv]     = Ab[arow * 64 + kbase];
        a[2 * vv + 1] = Ab[arow * 64 + kbase + 1];
        int kc = sub + ((vv < 4) ? (2 * vv) : (16 + 2 * (vv - 4))) + ka;
        b0[2 * vv]     = Bb[(niw + ncl) * 64 + kc];
        b0[2 * vv + 1] = Bb[(niw + ncl) * 64 + kc + 1];
        b1[2 * vv]     = Bb[(niw + 16 + ncl) * 64 + kc];
        b1[2 * vv + 1] = Bb[(niw + 16 + ncl) * 64 + kc + 1];
      }
      acc0 = __builtin_amdgcn_wmma_f32_16x16x32_f16(false, a, false, b0,
                                                    (short)0, acc0, false, false);
      acc1 = __builtin_amdgcn_wmma_f32_16x16x32_f16(false, a, false, b1,
                                                    (short)0, acc1, false, false);
    }
    if (more) storeAB(buf ^ 1);   // writes go to the inactive buffer
    __syncthreads();              // single barrier per K-step
    buf ^= 1;
  }

  const int nc0 = n0 + niw + ncl;
  const int nc1 = nc0 + 16;
#pragma unroll
  for (int r = 0; r < 8; ++r) {
    int m = m0 + miw + r + ((lane >> 4) << 3);
    if (nc0 < N) Dout[(size_t)m * N + nc0] = acc0[r];
    if (nc1 < N) Dout[(size_t)m * N + nc1] = acc1[r];
  }
}

// ---------------------------------------------------------------------------
// GEMM 2 (CHW 1x1 conv): per batch z, Out[n*HW+m] = sum_k Wt[n*K+k]*X[k*HW+m]
// Requirements: K % 64 == 0, N % 32 == 0, HW % 64 == 0 (true for call sites).
// Fully unguarded; same pipeline/double-buffer structure as gemm_xw_t.
// ---------------------------------------------------------------------------
__global__ __launch_bounds__(128)
void gemm_chw(const float* __restrict__ Wt, const float* __restrict__ X,
              float* __restrict__ Out, int N, int K, int HW,
              size_t xStride, size_t oStride) {
  __shared__ _Float16 As[2][32 * 64];   // 32 n-rows x 64 k (weights)
  __shared__ _Float16 Bs[2][64 * 64];   // 64 k-rows x 64 m (activations)
  const int tid = threadIdx.x, lane = tid & 31, wave = tid >> 5;
  const int m0 = blockIdx.x * 64;
  const int n0 = blockIdx.y * 32;
  const size_t xoff = (size_t)blockIdx.z * xStride;
  const size_t ooff = (size_t)blockIdx.z * oStride;
  const int niw = (wave >> 1) << 4;    // wave n offset (0/16)
  const int miw = (wave & 1) << 5;     // wave m offset (0/32)

  float4 ra[4];
  float4 rb[8];

  auto loadA = [&](int k0) {
#pragma unroll
    for (int j = 0; j < 4; ++j) {
      int i4 = tid + j * 128;
      int r = i4 >> 4, kq = (i4 & 15) << 2;
      ra[j] = *(const float4*)&Wt[(size_t)(n0 + r) * K + k0 + kq];
    }
  };
  auto loadB = [&](int k0) {
#pragma unroll
    for (int j = 0; j < 8; ++j) {
      int i4 = tid + j * 128;
      int kk = i4 >> 4, mq = (i4 & 15) << 2;
      rb[j] = *(const float4*)&X[xoff + (size_t)(k0 + kk) * HW + m0 + mq];
    }
  };
  auto storeAB = [&](int buf) {
#pragma unroll
    for (int j = 0; j < 4; ++j)
      *(h4*)&As[buf][(tid + j * 128) << 2] = cvt4(ra[j]);
#pragma unroll
    for (int j = 0; j < 8; ++j)
      *(h4*)&Bs[buf][(tid + j * 128) << 2] = cvt4(rb[j]);
  };

  v8f acc0 = {}, acc1 = {};
  const int arow = niw + (lane & 15);
  const int ka = (lane >> 4) << 3;
  const int kb = (lane >> 4) << 4;
  const int mcl = lane & 15;

  loadA(0); loadB(0);
  storeAB(0);
  __syncthreads();

  int buf = 0;
  for (int k0 = 0; k0 < K; k0 += 64) {
    const bool more = (k0 + 64) < K;
    if (more) {
      loadA(k0 + 64);
      loadB(k0 + 64);
    }
    const _Float16* Ab = As[buf];
    const _Float16* Bb = Bs[buf];
#pragma unroll
    for (int sub = 0; sub < 64; sub += 32) {
      v16h a, b0, b1;
#pragma unroll
      for (int vv = 0; vv < 8; ++vv) {
        int kbase = sub + ((vv < 4) ? (2 * vv) : (16 + 2 * (vv - 4))) + ka;
        a[2 * vv]     = Ab[arow * 64 + kbase];
        a[2 * vv + 1] = Ab[arow * 64 + kbase + 1];
        int kc = sub + ((vv < 4) ? (2 * vv) : (16 + 2 * (vv - 4))) + ka;   // B stored k-major: [k][m]
        b0[2 * vv]     = Bb[kc * 64 + miw + mcl];
        b0[2 * vv + 1] = Bb[(kc + 1) * 64 + miw + mcl];
        b1[2 * vv]     = Bb[kc * 64 + miw + 16 + mcl];
        b1[2 * vv + 1] = Bb[(kc + 1) * 64 + miw + 16 + mcl];
      }
      acc0 = __builtin_amdgcn_wmma_f32_16x16x32_f16(false, a, false, b0,
                                                    (short)0, acc0, false, false);
      acc1 = __builtin_amdgcn_wmma_f32_16x16x32_f16(false, a, false, b1,
                                                    (short)0, acc1, false, false);
    }
    if (more) storeAB(buf ^ 1);
    __syncthreads();
    buf ^= 1;
  }

  const int mc0 = m0 + miw + mcl;
  const int mc1 = mc0 + 16;
#pragma unroll
  for (int r = 0; r < 8; ++r) {
    int n = n0 + niw + r + ((lane >> 4) << 3);
    Out[ooff + (size_t)n * HW + mc0] = acc0[r];
    Out[ooff + (size_t)n * HW + mc1] = acc1[r];
  }
}

// ---------------------------------------------------------------------------
// Depthwise 3x3, pad 1, optional bias. One thread per output element.
// ---------------------------------------------------------------------------
__global__ __launch_bounds__(256)
void dwconv3x3(const float* __restrict__ in, const float* __restrict__ w,
               const float* __restrict__ bias, float* __restrict__ out,
               int C, int Hh, int Ww, int total) {
  int idx = blockIdx.x * blockDim.x + threadIdx.x;
  if (idx >= total) return;
  int x = idx % Ww;
  int y = (idx / Ww) % Hh;
  int c = (idx / (Ww * Hh)) % C;
  size_t plane = (size_t)(idx / (Ww * Hh)) * Hh * Ww;
  float acc = bias ? bias[c] : 0.0f;
#pragma unroll
  for (int ky = 0; ky < 3; ++ky) {
    int yy = y + ky - 1;
    if (yy < 0 || yy >= Hh) continue;
#pragma unroll
    for (int kx = 0; kx < 3; ++kx) {
      int xx = x + kx - 1;
      if (xx < 0 || xx >= Ww) continue;
      acc += w[c * 9 + ky * 3 + kx] * in[plane + (size_t)yy * Ww + xx];
    }
  }
  out[idx] = acc;
}

// adaptive avg pool 64x64 -> s x s
__global__ __launch_bounds__(256)
void pool_k(const float* __restrict__ in, float* __restrict__ out, int s, int total) {
  int idx = blockIdx.x * blockDim.x + threadIdx.x;
  if (idx >= total) return;
  int px = idx % s, py = (idx / s) % s;
  int bc = idx / (s * s);     // b*C + c
  int f = 64 / s;
  float acc = 0.0f;
  size_t base = (size_t)bc * HW_;
  for (int dy = 0; dy < 16; ++dy) {
    if (dy >= f) break;
    for (int dx = 0; dx < 16; ++dx) {
      if (dx >= f) break;
      acc += in[base + (size_t)(py * f + dy) * 64 + (px * f + dx)];
    }
  }
  out[idx] = acc / (float)(f * f);
}

// depthwise 3x3 + bias on pooled s x s map, writing straight into xm (b,L,128)
__global__ __launch_bounds__(256)
void msp_k(const float* __restrict__ in, const float* __restrict__ w,
           const float* __restrict__ bias, float* __restrict__ xm,
           int s, int l0, int total) {
  int idx = blockIdx.x * blockDim.x + threadIdx.x;
  if (idx >= total) return;
  int x = idx % s, y = (idx / s) % s;
  int c = (idx / (s * s)) % DIM_;
  int b = idx / (s * s * DIM_);
  size_t plane = ((size_t)b * DIM_ + c) * s * s;
  float acc = bias[c];
#pragma unroll
  for (int ky = 0; ky < 3; ++ky) {
    int yy = y + ky - 1;
    if (yy < 0 || yy >= s) continue;
#pragma unroll
    for (int kx = 0; kx < 3; ++kx) {
      int xx = x + kx - 1;
      if (xx < 0 || xx >= s) continue;
      acc += w[c * 9 + ky * 3 + kx] * in[plane + (size_t)yy * s + xx];
    }
  }
  xm[((size_t)b * L_ + l0 + y * s + x) * DIM_ + c] = acc;
}

// causal depthwise conv1d (k=4, left pad 3) + bias + SiLU: xr[:, :256] -> xc
__global__ __launch_bounds__(256)
void conv1d_silu_k(const float* __restrict__ xr, const float* __restrict__ w,
                   const float* __restrict__ bias, float* __restrict__ xc, int total) {
  int idx = blockIdx.x * blockDim.x + threadIdx.x;
  if (idx >= total) return;
  int c = idx % DIN_;
  int row = idx / DIN_;       // b*L + l
  int l = row % L_;
  float acc = bias[c];
#pragma unroll
  for (int t = 0; t < 4; ++t) {
    int li = l - 3 + t;
    if (li >= 0)
      acc += w[c * 4 + t] * xr[(size_t)(row - 3 + t) * 512 + c];
  }
  xc[idx] = silu_f(acc);
}

// delta = softplus(dt @ dt_proj.T + dt_bias);  dt = xdbl[:, :8]
__global__ __launch_bounds__(256)
void delta_k(const float* __restrict__ xdbl, const float* __restrict__ dtw,
             const float* __restrict__ dtb, float* __restrict__ delta, int total) {
  int idx = blockIdx.x * blockDim.x + threadIdx.x;
  if (idx >= total) return;
  int d = idx % DIN_;
  size_t row = idx / DIN_;
  float s = dtb[d];
#pragma unroll
  for (int r = 0; r < DTRANK_; ++r)
    s += xdbl[row * 40 + r] * dtw[d * DTRANK_ + r];
  delta[idx] = (s > 20.0f) ? s : log1pf(__expf(s));
}

// selective scan: lane-per-state (16 lanes per (b,d) sequence), shfl reduction
__global__ __launch_bounds__(256)
void scan_k(const float* __restrict__ delta, const float* __restrict__ xc,
            const float* __restrict__ xdbl, const float* __restrict__ A_log,
            const float* __restrict__ Dp, float* __restrict__ ybuf) {
  int t = blockIdx.x * blockDim.x + threadIdx.x;
  int g = t >> 4;           // sequence id: b*DIN + d
  int n = t & 15;           // state index
  if (g >= B_ * DIN_) return;
  int b = g >> 8;
  int d = g & 255;
  float A  = -__expf(A_log[d * DSTATE_ + n]);
  float Dd = Dp[d];
  float h = 0.0f;
  size_t base = (size_t)b * L_;
  for (int l = 0; l < L_; ++l) {
    size_t row = base + l;
    float da = delta[row * DIN_ + d];
    float u  = xc[row * DIN_ + d];
    float Bv = xdbl[row * 40 + DTRANK_ + n];
    float Cv = xdbl[row * 40 + DTRANK_ + DSTATE_ + n];
    h = h * __expf(da * A) + da * u * Bv;
    float p = h * Cv;
    p += __shfl_xor(p, 1, 16);
    p += __shfl_xor(p, 2, 16);
    p += __shfl_xor(p, 4, 16);
    p += __shfl_xor(p, 8, 16);
    if (n == 0) ybuf[row * DIN_ + d] = p + u * Dd;
  }
}

// y *= silu(res), res = xr[:, 256:512]
__global__ __launch_bounds__(256)
void gate_k(float* __restrict__ y, const float* __restrict__ xr, int total) {
  int idx = blockIdx.x * blockDim.x + threadIdx.x;
  if (idx >= total) return;
  int d = idx % DIN_;
  size_t row = idx / DIN_;
  float r = xr[row * 512 + DIN_ + d];
  y[idx] *= silu_f(r);
}

// split xm back into s x s maps, bilinear (half-pixel) upsample to 64x64,
// write channel-concatenated cat (b, 3*128, 64, 64)
__global__ __launch_bounds__(256)
void upsample_k(const float* __restrict__ xm, float* __restrict__ cat, int total) {
  int idx = blockIdx.x * blockDim.x + threadIdx.x;
  if (idx >= total) return;
  int x = idx & 63;
  int y = (idx >> 6) & 63;
  int c = (idx >> 12) & 127;
  int i = (idx >> 19) % 3;
  int b = idx / (3 * DIM_ * HW_);
  int s  = (i == 0) ? 8 : (i == 1) ? 16 : 32;
  int l0 = (i == 0) ? 0 : (i == 1) ? 64 : 320;
  float fs = (float)s / 64.0f;
  float fy = ((float)y + 0.5f) * fs - 0.5f;
  float fx = ((float)x + 0.5f) * fs - 0.5f;
  int y0 = (int)floorf(fy), x0 = (int)floorf(fx);
  float wy = fy - (float)y0, wx = fx - (float)x0;
  int y0c = min(max(y0, 0), s - 1), y1c = min(max(y0 + 1, 0), s - 1);
  int x0c = min(max(x0, 0), s - 1), x1c = min(max(x0 + 1, 0), s - 1);
  size_t base = ((size_t)b * L_ + l0) * DIM_ + c;
  float v00 = xm[base + (size_t)(y0c * s + x0c) * DIM_];
  float v01 = xm[base + (size_t)(y0c * s + x1c) * DIM_];
  float v10 = xm[base + (size_t)(y1c * s + x0c) * DIM_];
  float v11 = xm[base + (size_t)(y1c * s + x1c) * DIM_];
  float v = (1.0f - wy) * ((1.0f - wx) * v00 + wx * v01) +
            wy * ((1.0f - wx) * v10 + wx * v11);
  cat[((size_t)b * 3 * DIM_ + (i * DIM_ + c)) * HW_ + y * 64 + x] = v;
}

// ---------------------------------------------------------------------------
extern "C" void kernel_launch(void* const* d_in, const int* in_sizes, int n_in,
                              void* d_out, int out_size, void* d_ws, size_t ws_size,
                              hipStream_t stream) {
  (void)in_sizes; (void)n_in; (void)out_size; (void)ws_size;
  const float* x0       = (const float*)d_in[0];
  const float* conv_w   = (const float*)d_in[1];
  const float* dwconv_w = (const float*)d_in[2];
  const float* msp_w[3] = {(const float*)d_in[3], (const float*)d_in[4], (const float*)d_in[5]};
  const float* msp_b[3] = {(const float*)d_in[6], (const float*)d_in[7], (const float*)d_in[8]};
  const float* dw_r_w   = (const float*)d_in[9];
  const float* conv_r_w = (const float*)d_in[10];
  // blocks at 11 + 9*blk: in_proj, conv1d_w, conv1d_b, x_proj, dt_proj,
  //                       dt_bias, A_log, D, out_proj

  float* ws = (float*)d_ws;
  float* t0     = ws;                    // 4,194,304  conv1x1 out
  float* xbuf   = t0   + 4194304;        // 4,194,304  dwconv out
  float* xmA    = xbuf + 4194304;        // 1,376,256  (B,L,128)
  float* xmB    = xmA  + 1376256;        // 1,376,256
  float* xr     = xmB  + 1376256;        // 5,505,024  (B,L,512)
  float* xc     = xr   + 5505024;        // 2,752,512  (B,L,256)
  float* xdbl   = xc   + 2752512;        //   430,080  (B,L,40)
  float* dbuf   = xdbl + 430080;         // 2,752,512  delta
  float* ybuf   = dbuf + 2752512;        // 2,752,512
  float* pooled = ybuf + 2752512;        // 1,048,576
  // tail phase reuses dead regions:
  float* cat    = ws + 12582912;         // 12,582,912 (B,384,64,64) over mamba bufs
  float* dwr    = ws;                    // 12,582,912 over t0/xbuf/xm (dead)

  const int M = B_ * L_;                 // 10752 token rows

  // 1) conv1x1: t0[b,o,hw] = conv_w[o,:] . x0[b,:,hw]
  gemm_chw<<<dim3(HW_ / 64, DIM_ / 32, B_), 128, 0, stream>>>(
      conv_w, x0, t0, DIM_, DIM_, HW_, (size_t)DIM_ * HW_, (size_t)DIM_ * HW_);

  // 2) depthwise 3x3 (no bias)
  int total = B_ * DIM_ * HW_;
  dwconv3x3<<<(total + 255) / 256, 256, 0, stream>>>(
      t0, dwconv_w, nullptr, xbuf, DIM_, H_, W_, total);

  // 3) pyramid pooling -> msp depthwise conv -> xmA (b, l, c)
  const int scales[3] = {8, 16, 32};
  const int l0s[3]    = {0, 64, 320};
  for (int i = 0; i < 3; ++i) {
    int s = scales[i];
    int tp = B_ * DIM_ * s * s;
    pool_k<<<(tp + 255) / 256, 256, 0, stream>>>(xbuf, pooled, s, tp);
    msp_k<<<(tp + 255) / 256, 256, 0, stream>>>(pooled, msp_w[i], msp_b[i],
                                                xmA, s, l0s[i], tp);
  }

  // 4) two Mamba blocks
  const float* xm_in = xmA;
  float* xm_out = xmB;
  for (int blk = 0; blk < 2; ++blk) {
    const float* in_proj  = (const float*)d_in[11 + blk * 9 + 0];
    const float* conv1d_w = (const float*)d_in[11 + blk * 9 + 1];
    const float* conv1d_b = (const float*)d_in[11 + blk * 9 + 2];
    const float* x_proj   = (const float*)d_in[11 + blk * 9 + 3];
    const float* dt_proj  = (const float*)d_in[11 + blk * 9 + 4];
    const float* dt_bias  = (const float*)d_in[11 + blk * 9 + 5];
    const float* A_log    = (const float*)d_in[11 + blk * 9 + 6];
    const float* Dp       = (const float*)d_in[11 + blk * 9 + 7];
    const float* out_proj = (const float*)d_in[11 + blk * 9 + 8];

    // xr = xm @ in_proj.T   (M x 128) x (512 x 128)^T
    gemm_xw_t<<<dim3(512 / 64, M / 32), 128, 0, stream>>>(
        xm_in, in_proj, xr, M, 512, DIM_);

    int t1 = M * DIN_;
    conv1d_silu_k<<<(t1 + 255) / 256, 256, 0, stream>>>(xr, conv1d_w, conv1d_b, xc, t1);

    // xdbl = xc @ x_proj.T  (M x 256) x (40 x 256)^T  (N=40, guarded tile)
    gemm_xw_t<<<dim3(1, M / 32), 128, 0, stream>>>(xc, x_proj, xdbl, M, 40, DIN_);

    delta_k<<<(t1 + 255) / 256, 256, 0, stream>>>(xdbl, dt_proj, dt_bias, dbuf, t1);

    scan_k<<<(B_ * DIN_ * 16) / 256, 256, 0, stream>>>(dbuf, xc, xdbl, A_log, Dp, ybuf);

    gate_k<<<(t1 + 255) / 256, 256, 0, stream>>>(ybuf, xr, t1);

    // xm_out = gated @ out_proj.T  (M x 256) x (128 x 256)^T
    gemm_xw_t<<<dim3(DIM_ / 64, M / 32), 128, 0, stream>>>(
        ybuf, out_proj, xm_out, M, DIM_, DIN_);

    float* tmp = (float*)xm_in; xm_in = xm_out; xm_out = tmp;
  }

  // 5) bilinear upsample each scale to 64x64, channel-concat
  int tcat = B_ * 3 * DIM_ * HW_;
  upsample_k<<<(tcat + 255) / 256, 256, 0, stream>>>(xm_in, cat, tcat);

  // 6) depthwise 3x3 over 384 channels (no bias)
  dwconv3x3<<<(tcat + 255) / 256, 256, 0, stream>>>(
      cat, dw_r_w, nullptr, dwr, 3 * DIM_, H_, W_, tcat);

  // 7) final 1x1 conv 384 -> 128 into d_out
  gemm_chw<<<dim3(HW_ / 64, DIM_ / 32, B_), 128, 0, stream>>>(
      conv_r_w, dwr, (float*)d_out, DIM_, 3 * DIM_, HW_,
      (size_t)3 * DIM_ * HW_, (size_t)DIM_ * HW_);
}
